// HAN_13013750906996
// MI455X (gfx1250) — matrix operand order repaired and checked by reading, not compiled, not run on hardware
//
#include <hip/hip_runtime.h>
#include <hip/hip_fp16.h>

typedef __attribute__((ext_vector_type(16))) _Float16 v16h;
typedef __attribute__((ext_vector_type(8)))  float    v8f;

#define N   8192
#define IND 128
#define H   32
#define NW  (N/32)          // 256 mask words per row
#define LOG2E 1.44269504088896340736f

__device__ __forceinline__ float fast_rcp(float x)  { return __builtin_amdgcn_rcpf(x); }
__device__ __forceinline__ float fast_exp2(float x) { return __builtin_amdgcn_exp2f(x); }

// ---------- K1: pack int32 adjacency -> bitmask (256MB read once; 8MB out lives in L2) ----------
__global__ __launch_bounds__(256) void pack_kernel(const int* __restrict__ adj,
                                                   unsigned* __restrict__ bits) {
    size_t w = (size_t)blockIdx.x * 256 + threadIdx.x;       // word index, N*NW total
    const int4* p = (const int4*)adj + w * 8;                // 32 ints = 128B contiguous
    unsigned m = 0;
#pragma unroll
    for (int q = 0; q < 8; ++q) {
        int4 v = p[q];
        if (v.x) m |= 1u << (q*4+0);
        if (v.y) m |= 1u << (q*4+1);
        if (v.z) m |= 1u << (q*4+2);
        if (v.w) m |= 1u << (q*4+3);
    }
    bits[w] = m;
}

// ---------- K2: Wh = x @ W.T + b for both paths; emit f32 (for s) and f16 transposed (WMMA B) ----------
__global__ __launch_bounds__(256) void wh_kernel(const float* __restrict__ x,
                                                 const float* __restrict__ w0, const float* __restrict__ b0,
                                                 const float* __restrict__ w1, const float* __restrict__ b1,
                                                 float* __restrict__ whf, _Float16* __restrict__ whT) {
    int t    = blockIdx.x * 256 + threadIdx.x;   // 2 * N * H threads
    int path = t >> 18;
    int rem  = t & ((1 << 18) - 1);
    int i    = rem >> 5;
    int k    = rem & 31;
    const float* W  = path ? w1 : w0;
    const float* B  = path ? b1 : b0;
    const float4* xr = (const float4*)(x + (size_t)i * IND);
    const float4* wr = (const float4*)(W + (size_t)k * IND);
    float acc = B[k];
#pragma unroll
    for (int d = 0; d < IND/4; ++d) {
        float4 xv = xr[d], wv = wr[d];
        acc = fmaf(xv.x, wv.x, acc);
        acc = fmaf(xv.y, wv.y, acc);
        acc = fmaf(xv.z, wv.z, acc);
        acc = fmaf(xv.w, wv.w, acc);
    }
    whf[(size_t)path*(N*H) + (size_t)i*H + k] = acc;
    whT[(size_t)path*(H*N) + (size_t)k*N + i] = (_Float16)acc;  // transposed for B-operand loads
}

// ---------- K3: s_i = (Wh[i,:] . a) * log2(e)  (pre-scaled so hot loop uses raw v_exp_f32) ----------
// leaky_relu's max(v, 0.2v) commutes with the positive scale, so softmax weights are unchanged.
__global__ __launch_bounds__(256) void s_kernel(const float* __restrict__ whf,
                                                const float* __restrict__ a0,
                                                const float* __restrict__ a1,
                                                float* __restrict__ s) {
    int t    = blockIdx.x * 256 + threadIdx.x;  // 2*N threads
    int path = t >> 13;
    int i    = t & (N - 1);
    const float* a  = path ? a1 : a0;
    const float* wr = whf + (size_t)path*(N*H) + (size_t)i*H;
    float acc = 0.f;
#pragma unroll
    for (int k = 0; k < H; ++k) acc = fmaf(wr[k], a[k], acc);
    s[path*N + i] = acc * LOG2E;
}

// ---------- K4: global max of (scaled) s per path; zero scores ----------
__global__ __launch_bounds__(256) void smax_kernel(const float* __restrict__ s,
                                                   float* __restrict__ smax,
                                                   float* __restrict__ score) {
    __shared__ float red[256];
    int path = blockIdx.x;
    float m = -3.0e38f;
    for (int t = threadIdx.x; t < N; t += 256) m = fmaxf(m, s[path*N + t]);
    red[threadIdx.x] = m;
    __syncthreads();
    for (int o = 128; o > 0; o >>= 1) {
        if (threadIdx.x < o) red[threadIdx.x] = fmaxf(red[threadIdx.x], red[threadIdx.x + o]);
        __syncthreads();
    }
    if (threadIdx.x == 0) { smax[path] = red[0]; score[path] = 0.f; }
}

// ---------- K5 core sweep: 3 WMMAs/chunk: alpha@Wh (cols 0-15, 16-31) + rowsum via ones-B ----------
template <bool MASKED>
__device__ __forceinline__ void attn_sweep(const float* __restrict__ s_lds,
                                           const unsigned* __restrict__ brow,
                                           const _Float16* __restrict__ wt,
                                           float s_i, float m, int g, int nlo,
                                           v8f& c0, v8f& c1, v8f& c2) {
    // constant all-ones B: A(16x32) x ones(32x16) -> D[row][*] = row sum (f32 accumulate)
    v16h onesv;
#pragma unroll
    for (int i = 0; i < 16; ++i) onesv[i] = (_Float16)1.0f;

    const int gs = g << 3;  // 0 or 8

    for (int jb = 0; jb < NW; ++jb) {
        const int jbase = jb * 32;
        unsigned m0 = 0xffffffffu, m1 = 0xffffffffu;
        if (MASKED) {
            const unsigned mw = brow[jb];
            m0 = mw >> (gs << 1);       // bits for joff = g*16 + (0..7)      (VGPR r<4)
            m1 = mw >> ((gs << 1) + 8); // bits for joff = g*16 + 8 + (0..7)  (VGPR r>=4)
        }

        // Build A (16x32 f16 probabilities) in contiguous-k order (same as B):
        // VGPR r, half c:  K = 16*g + 2*r + c
        v16h a = {};
#pragma unroll
        for (int r = 0; r < 8; ++r) {
#pragma unroll
            for (int c = 0; c < 2; ++c) {
                const int sub  = ((r & 3) << 1) + c;             // 0..7, compile-time
                const int joff = (gs << 1) + ((r >> 2) << 3) + sub;
                const float v = s_i + s_lds[jbase + joff];
                const float e = fmaxf(v, 0.2f * v);              // leaky_relu(0.2), log2-domain
                float p = fast_exp2(e - m);                      // raw v_exp_f32
                if (MASKED) {
                    const unsigned mm = (r < 4) ? m0 : m1;
                    p = ((mm >> sub) & 1u) ? p : 0.f;            // constant shift after unroll
                }
                a[2*r + c] = (_Float16)p;
            }
        }

        // B (32x16 f16) = Whᵀ tile: lane n = nlo, 16 contiguous halves K = g*16 .. g*16+15
        const v16h b0 = *(const v16h*)(wt + (size_t)nlo        * N + jbase + g * 16);
        const v16h b1 = *(const v16h*)(wt + (size_t)(nlo + 16) * N + jbase + g * 16);

        c0 = __builtin_amdgcn_wmma_f32_16x16x32_f16(false, a, false, b0,    (short)0, c0, false, false);
        c1 = __builtin_amdgcn_wmma_f32_16x16x32_f16(false, a, false, b1,    (short)0, c1, false, false);
        c2 = __builtin_amdgcn_wmma_f32_16x16x32_f16(false, a, false, onesv, (short)0, c2, false, false);
    }
}

// ---------- K5: streaming masked-softmax attention, alpha@Wh via v_wmma_f32_16x16x32_f16 ----------
// grid (64, 2 paths), 256 threads = 8 waves; each wave owns a 16-row tile, sweeps 8192 cols in K=32 chunks.
__global__ __launch_bounds__(256) void attn_kernel(const float* __restrict__ s_all,
                                                   const float* __restrict__ smax_all,
                                                   const unsigned* __restrict__ bits,
                                                   const _Float16* __restrict__ whT,
                                                   float* __restrict__ z_all) {
    __shared__ float s_lds[N];                  // 32KB: whole (scaled) s vector, broadcast reads
    const int path = blockIdx.y;
    const float* s = s_all + path * N;
    for (int t = threadIdx.x; t < N; t += 256) s_lds[t] = s[t];
    __syncthreads();

    const int lane = threadIdx.x & 31;
    const int wave = threadIdx.x >> 5;
    const int g    = lane >> 4;                 // lane-group half
    const int nlo  = lane & 15;
    const int i0   = (blockIdx.x * 8 + wave) * 16;

    const float s_i  = s_lds[i0 + nlo];         // A-matrix row for this lane = nlo
    const float smax = smax_all[path];
    const float mvv  = s_i + smax;
    const float m    = fmaxf(mvv, 0.2f * mvv);  // exact row max (path1) / tight bound (path0)

    const _Float16* wt   = whT + (size_t)path * (H * N);
    const unsigned* brow = bits + (size_t)(i0 + nlo) * NW;

    v8f c0 = {}; v8f c1 = {}; v8f c2 = {};

    if (path == 0) attn_sweep<true >(s_lds, brow, wt, s_i, m, g, nlo, c0, c1, c2);
    else           attn_sweep<false>(s_lds, brow, wt, s_i, m, g, nlo, c0, c1, c2);

    // D layout: lane, VGPR r -> row = r + 8g, col = nlo (c0: cols 0-15, c1: cols 16-31).
    // c2[r] = full row sum of row r+8g (replicated over cols) -> no cross-lane exchange needed.
    float* z = z_all + (size_t)path * (N * H);
#pragma unroll
    for (int r = 0; r < 8; ++r) {
        const int   row  = (g << 3) + r;
        const float rsr  = c2[r];
        const float rinv = fast_rcp(rsr);        // v_rcp_f32
        const float d0   = c0[r], d1 = c1[r];
        const float z0   = (rsr > 0.f) ? fast_rcp(1.f + __expf(-d0 * rinv)) : 0.5f;
        const float z1v  = (rsr > 0.f) ? fast_rcp(1.f + __expf(-d1 * rinv)) : 0.5f;
        z[(size_t)(i0 + row) * H + nlo]      = z0;
        z[(size_t)(i0 + row) * H + 16 + nlo] = z1v;
    }
}

// ---------- K6: semantic path scores: sum_i tanh(Ws z_i + b) . q (block-reduced atomicAdd) ----------
__global__ __launch_bounds__(256) void score_kernel(const float* __restrict__ z_all,
                                                    const float* __restrict__ Wsw,
                                                    const float* __restrict__ Wsb,
                                                    const float* __restrict__ bsem,
                                                    const float* __restrict__ q,
                                                    float* __restrict__ score) {
    __shared__ float red[256];
    int t    = blockIdx.x * 256 + threadIdx.x;  // 2*N threads, path uniform per block
    int path = t >> 13;
    int i    = t & (N - 1);
    const float* zr = z_all + (size_t)path * (N * H) + (size_t)i * H;
    float zl[H];
#pragma unroll
    for (int k = 0; k < H; ++k) zl[k] = zr[k];
    float acc = 0.f;
#pragma unroll 4
    for (int k = 0; k < H; ++k) {
        float u = Wsb[k] + bsem[k];
        for (int j = 0; j < H; ++j) u = fmaf(Wsw[k * H + j], zl[j], u);
        acc = fmaf(tanhf(u), q[k], acc);
    }
    red[threadIdx.x] = acc;
    __syncthreads();
    for (int o = 128; o > 0; o >>= 1) {
        if (threadIdx.x < o) red[threadIdx.x] += red[threadIdx.x + o];
        __syncthreads();
    }
    if (threadIdx.x == 0) atomicAdd(&score[path], red[0]);
}

// ---------- K7: beta softmax, z_final, logits ----------
__global__ __launch_bounds__(256) void final_kernel(const float* __restrict__ z_all,
                                                    const float* __restrict__ score,
                                                    const float* __restrict__ clsw,
                                                    const float* __restrict__ clsb,
                                                    float* __restrict__ out) {
    int i = blockIdx.x * 256 + threadIdx.x;     // row
    const float inv_n = 1.f / (float)N;
    const float s0 = score[0] * inv_n, s1 = score[1] * inv_n;
    const float mx = fmaxf(s0, s1);
    const float e0 = __expf(s0 - mx), e1 = __expf(s1 - mx);
    const float einv = fast_rcp(e0 + e1);
    const float b0 = e0 * einv, b1 = e1 * einv;
    const float* z0 = z_all + (size_t)i * H;
    const float* z1 = z_all + (size_t)N * H + (size_t)i * H;
    float l0 = clsb[0], l1 = clsb[1];
#pragma unroll
    for (int k = 0; k < H; ++k) {
        const float zf = b0 * z0[k] + b1 * z1[k];
        out[(size_t)i * H + k] = zf;
        l0 = fmaf(zf, clsw[k],     l0);
        l1 = fmaf(zf, clsw[H + k], l1);
    }
    out[(size_t)N * H + i * 2 + 0] = l0;
    out[(size_t)N * H + i * 2 + 1] = l1;
}

extern "C" void kernel_launch(void* const* d_in, const int* in_sizes, int n_in,
                              void* d_out, int out_size, void* d_ws, size_t ws_size,
                              hipStream_t stream) {
    const float* x      = (const float*)d_in[0];
    const float* Wmgm_w = (const float*)d_in[1];
    const float* Wmgm_b = (const float*)d_in[2];
    const float* a_mgm  = (const float*)d_in[3];
    const float* Wm_w   = (const float*)d_in[4];
    const float* Wm_b   = (const float*)d_in[5];
    const float* a_m    = (const float*)d_in[6];
    const float* q      = (const float*)d_in[7];
    const float* Ws_w   = (const float*)d_in[8];
    const float* Ws_b   = (const float*)d_in[9];
    const float* b_sem  = (const float*)d_in[10];
    const float* cls_w  = (const float*)d_in[11];
    const float* cls_b  = (const float*)d_in[12];
    const int*   mgm_adj= (const int*)d_in[13];
    float* out = (float*)d_out;

    // workspace layout (bytes)
    char* ws = (char*)d_ws;
    unsigned* bits   = (unsigned*)(ws);                       // 8 MB
    float*    whf    = (float*)(ws + (8u  << 20));            // 2 MB  (2 x N x H f32)
    _Float16* whT    = (_Float16*)(ws + (10u << 20));         // 1 MB  (2 x H x N f16)
    float*    s      = (float*)(ws + (11u << 20));            // 64 KB (2 x N, scaled by log2e)
    float*    smax   = (float*)(ws + (11u << 20) + 65536);    // 2 f32
    float*    score  = (float*)(ws + (11u << 20) + 65536 + 64);
    float*    z      = (float*)(ws + (11u << 20) + 65536 + 128); // 2 MB (2 x N x H)

    pack_kernel <<<dim3((N * NW) / 256), dim3(256), 0, stream>>>(mgm_adj, bits);
    wh_kernel   <<<dim3((2 * N * H) / 256), dim3(256), 0, stream>>>(x, Wmgm_w, Wmgm_b, Wm_w, Wm_b, whf, whT);
    s_kernel    <<<dim3((2 * N) / 256), dim3(256), 0, stream>>>(whf, a_mgm, a_m, s);
    smax_kernel <<<dim3(2), dim3(256), 0, stream>>>(s, smax, score);
    attn_kernel <<<dim3(64, 2), dim3(256), 0, stream>>>(s, smax, bits, whT, z);
    score_kernel<<<dim3((2 * N) / 256), dim3(256), 0, stream>>>(z, Ws_w, Ws_b, b_sem, q, score);
    final_kernel<<<dim3(N / 256), dim3(256), 0, stream>>>(z, score, cls_w, cls_b, out);
}
